// NYLONModel_89790586290846
// MI455X (gfx1250) — matrix operand order repaired and checked by reading, not compiled, not run on hardware
//
#include <hip/hip_runtime.h>
#include <hip/hip_bf16.h>
#include <math.h>

// ---------------- problem constants ----------------
constexpr int B_    = 512;
constexpr int S_    = 17;
constexpr int V_    = 40002;
constexpr int E_    = 1024;
constexpr int H_    = 16;
constexpr int L_    = 12;
constexpr int F_    = 4096;
constexpr int DH_   = 64;       // E/H
constexpr int M_    = B_ * S_;  // 8704 rows of the activation matrix
constexpr int NREL_ = 500;
constexpr float BIG_ = 1000000.0f;

typedef __bf16 bf16;
typedef __attribute__((ext_vector_type(16))) __bf16 v16bf;
typedef __attribute__((ext_vector_type(8)))  float  v8f;

// ---------------- small device helpers ----------------
__device__ __forceinline__ float gelu_exact(float x) {
    return 0.5f * x * (1.0f + erff(x * 0.70710678118654752f));
}

__device__ __forceinline__ float block_reduce_sum(float v, float* red) {
    int tid = threadIdx.x;
    red[tid] = v; __syncthreads();
    for (int s = blockDim.x >> 1; s > 0; s >>= 1) {
        if (tid < s) red[tid] += red[tid + s];
        __syncthreads();
    }
    float r = red[0]; __syncthreads();
    return r;
}

__device__ __forceinline__ float block_reduce_max(float v, float* red) {
    int tid = threadIdx.x;
    red[tid] = v; __syncthreads();
    for (int s = blockDim.x >> 1; s > 0; s >>= 1) {
        if (tid < s) red[tid] = fmaxf(red[tid], red[tid + s]);
        __syncthreads();
    }
    float r = red[0]; __syncthreads();
    return r;
}

// ti value from the candidate-mask construction in the reference
__device__ __forceinline__ float ti_val(int col, float mt) {
    float t = (col < 2) ? -1.0f : ((col < 2 + NREL_) ? -mt : mt);
    return t > 0.0f ? t : 0.0f;
}

// ---------------- conversion kernels ----------------
// src fp32 [Kd][Nd] row-major -> dst bf16 [Nd][Kd]  (weight pre-transpose)
__global__ void convert_transpose_bf16(const float* __restrict__ src,
                                       bf16* __restrict__ dst,
                                       int Kd, int Nd, int total) {
    int idx = blockIdx.x * blockDim.x + threadIdx.x;
    if (idx < total) {
        int k = idx / Nd;
        int n = idx - k * Nd;
        dst[(long)n * Kd + k] = (bf16)src[idx];
    }
}

__global__ void convert_bf16(const float* __restrict__ src,
                             bf16* __restrict__ dst, int total) {
    int idx = blockIdx.x * blockDim.x + threadIdx.x;
    if (idx < total) dst[idx] = (bf16)src[idx];
}

// ---------------- embedding gather + LayerNorm ----------------
__global__ __launch_bounds__(256)
void embed_ln_kernel(const int* __restrict__ ids,
                     const float* __restrict__ emb,
                     const float* __restrict__ g,
                     const float* __restrict__ bp,
                     float* __restrict__ xout,
                     bf16* __restrict__ hbout) {
    int tok = blockIdx.x;
    int tid = threadIdx.x;
    __shared__ float buf[E_];
    __shared__ float red[256];
    const float* row = emb + (long)ids[tok] * E_;
    float s = 0.f;
    for (int i = tid; i < E_; i += 256) { float v = row[i]; buf[i] = v; s += v; }
    s = block_reduce_sum(s, red);
    float mean = s * (1.0f / E_);
    float s2 = 0.f;
    for (int i = tid; i < E_; i += 256) { float d = buf[i] - mean; s2 += d * d; }
    s2 = block_reduce_sum(s2, red);
    float rstd = rsqrtf(s2 * (1.0f / E_) + 1e-12f);
    for (int i = tid; i < E_; i += 256) {
        float o = (buf[i] - mean) * rstd * g[i] + bp[i];
        xout[(long)tok * E_ + i]  = o;
        hbout[(long)tok * E_ + i] = (bf16)o;
    }
}

// ---------------- residual + LayerNorm (in-place on x) ----------------
__global__ __launch_bounds__(256)
void res_ln_kernel(float* __restrict__ x,
                   const float* __restrict__ y,
                   const float* __restrict__ g,
                   const float* __restrict__ bp,
                   bf16* __restrict__ hbout) {
    int tok = blockIdx.x;
    int tid = threadIdx.x;
    __shared__ float buf[E_];
    __shared__ float red[256];
    float s = 0.f;
    for (int i = tid; i < E_; i += 256) {
        float v = x[(long)tok * E_ + i] + y[(long)tok * E_ + i];
        buf[i] = v; s += v;
    }
    s = block_reduce_sum(s, red);
    float mean = s * (1.0f / E_);
    float s2 = 0.f;
    for (int i = tid; i < E_; i += 256) { float d = buf[i] - mean; s2 += d * d; }
    s2 = block_reduce_sum(s2, red);
    float rstd = rsqrtf(s2 * (1.0f / E_) + 1e-12f);
    for (int i = tid; i < E_; i += 256) {
        float o = (buf[i] - mean) * rstd * g[i] + bp[i];
        x[(long)tok * E_ + i]     = o;
        hbout[(long)tok * E_ + i] = (bf16)o;
    }
}

// ---------------- edge gather ek/ev ----------------
__global__ void edge_gather_kernel(const int* __restrict__ labels,
                                   const float* __restrict__ ekt,
                                   const float* __restrict__ evt,
                                   float* __restrict__ ek,
                                   float* __restrict__ ev) {
    int idx = blockIdx.x * blockDim.x + threadIdx.x;
    if (idx < S_ * S_ * DH_) {
        int p = idx / DH_;
        int d = idx - p * DH_;
        int lbl = labels[p];
        float m = (lbl > 0) ? 1.0f : 0.0f;  // labels are non-negative
        ek[idx] = ekt[lbl * DH_ + d] * m;
        ev[idx] = evt[lbl * DH_ + d] * m;
    }
}

// ---------------- attention bias from input_mask ----------------
__global__ void attn_bias_kernel(const float* __restrict__ mask,
                                 float* __restrict__ bias) {
    int idx = blockIdx.x * blockDim.x + threadIdx.x;
    if (idx < B_ * S_ * S_) {
        int b = idx / (S_ * S_);
        int r = (idx / S_) % S_;
        int c = idx % S_;
        bias[idx] = BIG_ * (mask[b * S_ + r] * mask[b * S_ + c] - 1.0f);
    }
}

// ---------------- WMMA bf16 GEMM: C = A[M,K] * Bt[N,K]^T ----------------
// One wave computes a 32x64 macro-tile (MT=2 x NT=4 WMMA tiles): 8 WMMAs per
// 12 b128 loads per K-step (2.7x the arithmetic intensity of a 16x16 tile),
// keeping the kernel matrix-bound instead of L2-load-bound.
// EPI 0: outF = (acc + bias[col]) * scale
// EPI 1: outH = bf16(gelu((acc + bias[col]) * scale))
// EPI 2: outF = acc + bias[col] + BIG*(ti(col, mask_type[row]) - 1)  (logits)
union FragU { v16bf f; uint4 q[2]; };

constexpr int MT_ = 2;   // M tiles per wave (32 rows)
constexpr int NT_ = 4;   // N tiles per wave (64 cols)

template <int EPI>
__global__ __launch_bounds__(32)
void gemm_wmma_bf16(const bf16* __restrict__ A,
                    const bf16* __restrict__ Bt,
                    const float* __restrict__ bias,
                    float scale,
                    float* __restrict__ outF,
                    bf16* __restrict__ outH,
                    const float* __restrict__ mask_type,
                    int Mtot, int Ntot, int Kd) {
    const int lane = threadIdx.x & 31;
    const int half = lane >> 4;     // 0 or 1
    const int lrow = lane & 15;
    const int n0 = blockIdx.x * (16 * NT_);
    const int m0 = blockIdx.y * (16 * MT_);

    // A fragment: lane (half h) holds its row, K = kb + h*8..+7 and kb+16+h*8..+7
    const bf16* aBase[MT_];
    #pragma unroll
    for (int mt = 0; mt < MT_; ++mt)
        aBase[mt] = A + (long)(m0 + mt * 16 + lrow) * Kd + half * 8;

    // B fragment: lane holds one column, K = kb + h*8..+7 and kb+16+h*8..+7 (two b128 chunks)
    const bf16* bBase[NT_];
    int colv[NT_], nclv[NT_];
    #pragma unroll
    for (int nt = 0; nt < NT_; ++nt) {
        int col = n0 + nt * 16 + lrow;
        int ncl = (col < Ntot) ? col : (Ntot - 1);   // clamp edge vocab tile
        colv[nt] = col; nclv[nt] = ncl;
        bBase[nt] = Bt + (long)ncl * Kd + half * 8;
    }

    v8f acc[MT_][NT_];
    {
        v8f z = {};
        #pragma unroll
        for (int mt = 0; mt < MT_; ++mt)
            #pragma unroll
            for (int nt = 0; nt < NT_; ++nt) acc[mt][nt] = z;
    }

    for (int kb = 0; kb < Kd; kb += 32) {
        FragU a[MT_], b[NT_];
        #pragma unroll
        for (int mt = 0; mt < MT_; ++mt) {
            a[mt].q[0] = *(const uint4*)(aBase[mt] + kb);
            a[mt].q[1] = *(const uint4*)(aBase[mt] + kb + 16);
        }
        #pragma unroll
        for (int nt = 0; nt < NT_; ++nt) {
            b[nt].q[0] = *(const uint4*)(bBase[nt] + kb);
            b[nt].q[1] = *(const uint4*)(bBase[nt] + kb + 16);
        }
        if (kb + 32 < Kd) {
            #pragma unroll
            for (int mt = 0; mt < MT_; ++mt)
                __builtin_prefetch(aBase[mt] + kb + 32, 0, 0);  // global_prefetch_b8
            #pragma unroll
            for (int nt = 0; nt < NT_; ++nt)
                __builtin_prefetch(bBase[nt] + kb + 32, 0, 0);
        }
        #pragma unroll
        for (int mt = 0; mt < MT_; ++mt)
            #pragma unroll
            for (int nt = 0; nt < NT_; ++nt)
                acc[mt][nt] = __builtin_amdgcn_wmma_f32_16x16x32_bf16(
                        /*neg_a=*/false, a[mt].f, /*neg_b=*/false, b[nt].f,
                        /*c_mod=*/(short)0, acc[mt][nt],
                        /*reuse_a=*/false, /*reuse_b=*/false);
    }

    #pragma unroll
    for (int nt = 0; nt < NT_; ++nt) {
        const float bcolv = bias ? bias[nclv[nt]] : 0.0f;
        const int col = colv[nt];
        #pragma unroll
        for (int mt = 0; mt < MT_; ++mt) {
            #pragma unroll
            for (int r = 0; r < 8; ++r) {
                int row = m0 + mt * 16 + half * 8 + r;
                float v = acc[mt][nt][r];
                if (EPI == 0) {
                    outF[(long)row * Ntot + col] = (v + bcolv) * scale;
                } else if (EPI == 1) {
                    float g = gelu_exact((v + bcolv) * scale);
                    outH[(long)row * Ntot + col] = (bf16)g;
                } else { // EPI == 2 : vocab logits + candidate mask
                    if (col < Ntot) {
                        float mtv = mask_type[row];
                        outF[(long)row * Ntot + col] =
                            v + bcolv + BIG_ * (ti_val(col, mtv) - 1.0f);
                    }
                }
            }
        }
    }
}

// ---------------- relation-aware attention (S=17, DH=64) ----------------
__global__ __launch_bounds__(256)
void attn_kernel(const float* __restrict__ q,
                 const float* __restrict__ k,
                 const float* __restrict__ v,
                 const float* __restrict__ ek,
                 const float* __restrict__ ev,
                 const float* __restrict__ bias,
                 bf16* __restrict__ ctx) {
    const int bh = blockIdx.x;
    const int b  = bh / H_;
    const int h  = bh - b * H_;
    const int tid = threadIdx.x;

    __shared__ float qs[S_ * DH_];
    __shared__ float ks[S_ * DH_];
    __shared__ float vs[S_ * DH_];
    __shared__ float sc[S_ * S_];

    const long base = ((long)b * S_) * E_ + h * DH_;
    for (int idx = tid; idx < S_ * DH_; idx += 256) {
        int i = idx >> 6, d = idx & 63;
        long off = base + (long)i * E_ + d;
        qs[idx] = q[off]; ks[idx] = k[off]; vs[idx] = v[off];
    }
    __syncthreads();

    for (int t = tid; t < S_ * S_; t += 256) {
        int i = t / S_, j = t - i * S_;
        const float* ekp = ek + (i * S_ + j) * DH_;
        float s = bias[((long)b * S_ + i) * S_ + j];
        #pragma unroll 8
        for (int d = 0; d < DH_; ++d)
            s += qs[i * DH_ + d] * (ks[j * DH_ + d] + ekp[d]);
        sc[t] = s;
    }
    __syncthreads();

    if (tid < S_) {
        int i = tid;
        float mx = -1e30f;
        for (int j = 0; j < S_; ++j) mx = fmaxf(mx, sc[i * S_ + j]);
        float sm = 0.f;
        for (int j = 0; j < S_; ++j) {
            float e = __expf(sc[i * S_ + j] - mx);
            sc[i * S_ + j] = e; sm += e;
        }
        float inv = 1.0f / sm;
        for (int j = 0; j < S_; ++j) sc[i * S_ + j] *= inv;
    }
    __syncthreads();

    for (int idx = tid; idx < S_ * DH_; idx += 256) {
        int i = idx >> 6, d = idx & 63;
        float acc = 0.f;
        #pragma unroll
        for (int j = 0; j < S_; ++j)
            acc += sc[i * S_ + j] * (vs[j * DH_ + d] + ev[(i * S_ + j) * DH_ + d]);
        ctx[base + (long)i * E_ + d] = (bf16)acc;
    }
}

// ---------------- masked-position gather + gelu + LN (eps 1e-7) ----------------
__global__ __launch_bounds__(256)
void mask_gelu_ln_kernel(const float* __restrict__ enc,
                         const int* __restrict__ mask_pos,
                         const float* __restrict__ g,
                         const float* __restrict__ bp,
                         bf16* __restrict__ hm) {
    int b = blockIdx.x;
    int tid = threadIdx.x;
    __shared__ float buf[E_];
    __shared__ float red[256];
    int pos = mask_pos[b];
    const float* row = enc + ((long)b * S_ + pos) * E_;
    float s = 0.f;
    for (int i = tid; i < E_; i += 256) { float v = gelu_exact(row[i]); buf[i] = v; s += v; }
    s = block_reduce_sum(s, red);
    float mean = s * (1.0f / E_);
    float s2 = 0.f;
    for (int i = tid; i < E_; i += 256) { float d = buf[i] - mean; s2 += d * d; }
    s2 = block_reduce_sum(s2, red);
    float rstd = rsqrtf(s2 * (1.0f / E_) + 1e-7f);
    for (int i = tid; i < E_; i += 256)
        hm[(long)b * E_ + i] = (bf16)((buf[i] - mean) * rstd * g[i] + bp[i]);
}

// ---------------- per-row soft-label cross entropy ----------------
__global__ __launch_bounds__(256)
void loss_row_kernel(const float* __restrict__ logits,
                     const int* __restrict__ mask_label,
                     const float* __restrict__ mask_type,
                     const float* __restrict__ conf,
                     float* __restrict__ rowloss) {
    int b = blockIdx.x;
    int tid = threadIdx.x;
    __shared__ float red[256];
    const float* row = logits + (long)b * V_;
    float mt = mask_type[b];
    int label = mask_label[b];

    float mx = -1e30f;
    for (int j = tid; j < V_; j += 256) mx = fmaxf(mx, row[j]);
    mx = block_reduce_max(mx, red);

    float se = 0.f;
    for (int j = tid; j < V_; j += 256) se += __expf(row[j] - mx);
    se = block_reduce_sum(se, red);
    float lse = logf(se);

    // closed-form sum of ti over the vocab
    float tin = fmaxf(-mt, 0.f), tip = fmaxf(mt, 0.f);
    float sum_ti   = (float)NREL_ * tin + (float)(V_ - NREL_ - 2) * tip;
    float num_cand = sum_ti - 1.0f;          // sum(ti - onehot)
    float sl = ((1.0f + mt) * 0.9f + (1.0f - mt) * 0.9f) * 0.5f;
    float w  = (1.0f - sl) / num_cand;

    float dot = 0.f;
    for (int j = tid; j < V_; j += 256) {
        float ti = ti_val(j, mt);
        float oh = (j == label) ? 1.0f : 0.0f;
        float soft = sl * oh + w * (ti - oh);
        dot += soft * (row[j] - mx);
    }
    dot = block_reduce_sum(dot, red);

    if (tid == 0) rowloss[b] = (lse - dot) * conf[b];   // sum(soft)==1
}

__global__ __launch_bounds__(256)
void loss_reduce_kernel(const float* __restrict__ rowloss, float* __restrict__ out) {
    __shared__ float red[256];
    int tid = threadIdx.x;
    float s = 0.f;
    for (int i = tid; i < B_; i += 256) s += rowloss[i];
    s = block_reduce_sum(s, red);
    if (tid == 0) out[0] = s * (1.0f / B_);
}

// ---------------- host launcher ----------------
extern "C" void kernel_launch(void* const* d_in, const int* in_sizes, int n_in,
                              void* d_out, int out_size, void* d_ws, size_t ws_size,
                              hipStream_t stream) {
    const int*   input_ids   = (const int*)  d_in[0];
    const float* input_mask  = (const float*)d_in[1];
    const int*   edge_labels = (const int*)  d_in[2];
    const int*   mask_pos    = (const int*)  d_in[3];
    const int*   mask_label  = (const int*)  d_in[4];
    const float* mask_type   = (const float*)d_in[5];
    const float* confidence  = (const float*)d_in[6];
    const float* node_emb    = (const float*)d_in[7];
    const float* edge_k      = (const float*)d_in[8];
    const float* edge_v      = (const float*)d_in[9];
    const float* ln1_g       = (const float*)d_in[10];
    const float* ln1_b       = (const float*)d_in[11];
    const float* Wq = (const float*)d_in[12]; const float* bq = (const float*)d_in[13];
    const float* Wk = (const float*)d_in[14]; const float* bk = (const float*)d_in[15];
    const float* Wv = (const float*)d_in[16]; const float* bv = (const float*)d_in[17];
    const float* Wo = (const float*)d_in[18]; const float* bo = (const float*)d_in[19];
    const float* lnA_g = (const float*)d_in[20]; const float* lnA_b = (const float*)d_in[21];
    const float* W1 = (const float*)d_in[22]; const float* b1 = (const float*)d_in[23];
    const float* W2 = (const float*)d_in[24]; const float* b2 = (const float*)d_in[25];
    const float* lnF_g = (const float*)d_in[26]; const float* lnF_b = (const float*)d_in[27];
    const float* ln2_g = (const float*)d_in[28]; const float* ln2_b = (const float*)d_in[29];
    const float* fc2_bias = (const float*)d_in[30];

    char* ws = (char*)d_ws;
    size_t off = 0;
    auto alloc = [&](size_t bytes) -> void* {
        void* p = ws + off;
        off += (bytes + 255) & ~(size_t)255;
        return p;
    };
    float* x    = (float*)alloc((size_t)M_ * E_ * 4);   // residual stream fp32
    bf16*  hb   = (bf16*) alloc((size_t)M_ * E_ * 2);   // activation bf16 (GEMM A)
    float* qb   = (float*)alloc((size_t)M_ * E_ * 4);
    float* kb   = (float*)alloc((size_t)M_ * E_ * 4);
    float* vb   = (float*)alloc((size_t)M_ * E_ * 4);
    bf16*  ctx  = (bf16*) alloc((size_t)M_ * E_ * 2);
    float* yb   = (float*)alloc((size_t)M_ * E_ * 4);
    bf16*  h1   = (bf16*) alloc((size_t)M_ * F_ * 2);
    bf16*  wt   = (bf16*) alloc((size_t)E_ * F_ * 2);   // transposed weight buffer
    bf16*  neb  = (bf16*) alloc((size_t)V_ * E_ * 2);   // node_emb in bf16 (already [N][K])
    float* ekw  = (float*)alloc((size_t)S_ * S_ * DH_ * 4);
    float* evw  = (float*)alloc((size_t)S_ * S_ * DH_ * 4);
    float* biasb= (float*)alloc((size_t)B_ * S_ * S_ * 4);
    bf16*  hm   = (bf16*) alloc((size_t)B_ * E_ * 2);
    float* rowl = (float*)alloc((size_t)B_ * 4);
    (void)ws_size; (void)in_sizes; (void)n_in; (void)out_size;

    // 1. embedding + LN1
    embed_ln_kernel<<<M_, 256, 0, stream>>>(input_ids, node_emb, ln1_g, ln1_b, x, hb);
    // 2. edge gathers + attention bias
    {
        int n = S_ * S_ * DH_;
        edge_gather_kernel<<<(n + 255) / 256, 256, 0, stream>>>(edge_labels, edge_k, edge_v, ekw, evw);
        int nb = B_ * S_ * S_;
        attn_bias_kernel<<<(nb + 255) / 256, 256, 0, stream>>>(input_mask, biasb);
    }
    // 3. node_emb -> bf16 (logits B operand, already [V][E] = [N][K])
    {
        int n = V_ * E_;
        convert_bf16<<<(n + 255) / 256, 256, 0, stream>>>(node_emb, neb, n);
    }

    // 32x64 macro-tiles per wave
    const dim3 gEE(E_ / 64, M_ / 32);   // N=1024 -> 16 x 272
    const dim3 gEF(F_ / 64, M_ / 32);   // N=4096 -> 64 x 272
    const int  tEE = E_ * E_, tEF = E_ * F_;

    for (int l = 0; l < L_; ++l) {
        const float* wq = Wq + (size_t)l * E_ * E_;
        const float* wk = Wk + (size_t)l * E_ * E_;
        const float* wv = Wv + (size_t)l * E_ * E_;
        const float* wo = Wo + (size_t)l * E_ * E_;
        const float* w1 = W1 + (size_t)l * E_ * F_;
        const float* w2 = W2 + (size_t)l * F_ * E_;

        // Q = (hb @ Wq + bq) * DH^-0.5
        convert_transpose_bf16<<<(tEE + 255) / 256, 256, 0, stream>>>(wq, wt, E_, E_, tEE);
        gemm_wmma_bf16<0><<<gEE, 32, 0, stream>>>(hb, wt, bq + (size_t)l * E_, 0.125f,
                                                  qb, nullptr, nullptr, M_, E_, E_);
        // K
        convert_transpose_bf16<<<(tEE + 255) / 256, 256, 0, stream>>>(wk, wt, E_, E_, tEE);
        gemm_wmma_bf16<0><<<gEE, 32, 0, stream>>>(hb, wt, bk + (size_t)l * E_, 1.0f,
                                                  kb, nullptr, nullptr, M_, E_, E_);
        // V
        convert_transpose_bf16<<<(tEE + 255) / 256, 256, 0, stream>>>(wv, wt, E_, E_, tEE);
        gemm_wmma_bf16<0><<<gEE, 32, 0, stream>>>(hb, wt, bv + (size_t)l * E_, 1.0f,
                                                  vb, nullptr, nullptr, M_, E_, E_);
        // attention with edge bias -> ctx (bf16)
        attn_kernel<<<B_ * H_, 256, 0, stream>>>(qb, kb, vb, ekw, evw, biasb, ctx);
        // y = ctx @ Wo + bo
        convert_transpose_bf16<<<(tEE + 255) / 256, 256, 0, stream>>>(wo, wt, E_, E_, tEE);
        gemm_wmma_bf16<0><<<gEE, 32, 0, stream>>>(ctx, wt, bo + (size_t)l * E_, 1.0f,
                                                  yb, nullptr, nullptr, M_, E_, E_);
        // h = LN(h + y)
        res_ln_kernel<<<M_, 256, 0, stream>>>(x, yb, lnA_g + (size_t)l * E_,
                                              lnA_b + (size_t)l * E_, hb);
        // h1 = gelu(h @ W1 + b1)  (bf16 out)
        convert_transpose_bf16<<<(tEF + 255) / 256, 256, 0, stream>>>(w1, wt, E_, F_, tEF);
        gemm_wmma_bf16<1><<<gEF, 32, 0, stream>>>(hb, wt, b1 + (size_t)l * F_, 1.0f,
                                                  nullptr, h1, nullptr, M_, F_, E_);
        // y = h1 @ W2 + b2
        convert_transpose_bf16<<<(tEF + 255) / 256, 256, 0, stream>>>(w2, wt, F_, E_, tEF);
        gemm_wmma_bf16<0><<<gEE, 32, 0, stream>>>(h1, wt, b2 + (size_t)l * E_, 1.0f,
                                                  yb, nullptr, nullptr, M_, E_, F_);
        // h = LN(h + y)
        res_ln_kernel<<<M_, 256, 0, stream>>>(x, yb, lnF_g + (size_t)l * E_,
                                              lnF_b + (size_t)l * E_, hb);
    }

    // head: gather masked positions, gelu, LN(eps 1e-7)
    mask_gelu_ln_kernel<<<B_, 256, 0, stream>>>(x, mask_pos, ln2_g, ln2_b, hm);

    // logits = hm @ node_emb^T + fc2_bias + BIG*(ti-1)   -> d_out[1..]
    float* logits = (float*)d_out + 1;
    dim3 gLog((V_ + 63) / 64, B_ / 32);   // 626 x 16
    gemm_wmma_bf16<2><<<gLog, 32, 0, stream>>>(hm, neb, fc2_bias, 1.0f,
                                               logits, nullptr, mask_type, B_, V_, E_);

    // loss -> d_out[0] (deterministic two-stage reduction)
    loss_row_kernel<<<B_, 256, 0, stream>>>(logits, mask_label, mask_type, confidence, rowl);
    loss_reduce_kernel<<<1, 256, 0, stream>>>(rowl, (float*)d_out);
}
